// MultiBarrelModel_50208167690760
// MI455X (gfx1250) — compile-verified, matrix operand order checked by reading, not run on hardware
//
#include <hip/hip_runtime.h>
#include <hip/hip_bf16.h>

// MI455X / gfx1250 multi-barrel spiking RNN.
// Latency-bound recurrence: one workgroup per barrel (39), batch=64 as GEMM M dim,
// 8 wave32s each own a 16x16 tile of every 64x32 pop-step GEMM (v_wmma_f32_16x16x32_f16).
// LIF state lives in registers (pop loop fully unrolled); Xt tiles staged via the
// Tensor Data Mover (tensor_load_to_lds + s_wait_tensorcnt) when available.

typedef __attribute__((ext_vector_type(16))) _Float16 v16h;
typedef __attribute__((ext_vector_type(8)))  float    v8f;
typedef __attribute__((ext_vector_type(2)))  _Float16 h2;

#define NB_    39
#define NPOP_  8
#define PSZ_   32
#define T_     64
#define B_     64
#define C_     64
#define CATD_  256   // NPOP_*PSZ_
#define NCONN_ 37

// connectivity (constexpr -> fully folded once the pop loop is unrolled)
constexpr int kSRC[NCONN_] = {
    0,1,2,3,4,        // pop0
    0,1,2,3,4,        // pop1
    1,2,3,5,6,7,      // pop2
    1,2,3,5,6,7,      // pop3
    0,4,6,7,          // pop4
    1,5,7,            // pop5
    2,3,4,5,6,7,      // pop6
    6,7               // pop7
};
constexpr int kNSRC[NPOP_] = {5,5,6,6,4,3,6,2};
constexpr int kOFF[NPOP_]  = {0,5,10,16,22,26,29,35};

// ---------------- TDM availability -------------------------------------------
#if defined(__has_builtin)
#  if __has_builtin(__builtin_amdgcn_tensor_load_to_lds) && __has_builtin(__builtin_amdgcn_s_wait_tensorcnt)
#    define HAVE_TDM 1
#  else
#    define HAVE_TDM 0
#  endif
#else
#  define HAVE_TDM 0
#endif

typedef __attribute__((ext_vector_type(4))) unsigned int v4u;
typedef __attribute__((ext_vector_type(8))) int          v8i;
typedef __attribute__((ext_vector_type(4))) int          v4i;

#if HAVE_TDM
// 1-D TDM tile: nelem f16 contiguous from global -> LDS (see D# layout, ISA ch.8)
__device__ __forceinline__ void tdm_load_1d(unsigned lds_off, const _Float16* gsrc, int nelem) {
    unsigned long long ga = (unsigned long long)gsrc;
    v4u g0;
    g0[0] = 1u;                                              // count=1 (user descriptor)
    g0[1] = lds_off;                                         // lds_addr (bytes)
    g0[2] = (unsigned)(ga & 0xffffffffull);                  // global_addr lo
    g0[3] = (unsigned)((ga >> 32) & 0x01ffffffull) | (2u << 30); // global_addr hi | type=2
    v8i g1;
    g1[0] = (int)(1u << 16);                                 // data_size=1 -> 2 bytes/elem
    g1[1] = (int)(((unsigned)nelem & 0xffffu) << 16);        // tensor_dim0 lo16
    g1[2] = (int)((((unsigned)nelem >> 16) & 0xffffu) | (1u << 16)); // dim0 hi | tensor_dim1=1
    g1[3] = (int)(((unsigned)nelem & 0xffffu) << 16);        // tile_dim0 = nelem
    g1[4] = 0;                                               // tile_dim1/2 unused
    g1[5] = nelem;                                           // tensor_dim0_stride lo32
    g1[6] = 0;
    g1[7] = 0;
    v4i gz = {0, 0, 0, 0};
#if __clang_major__ >= 23
    v8i gz8 = {0, 0, 0, 0, 0, 0, 0, 0};
    __builtin_amdgcn_tensor_load_to_lds(g0, g1, gz, gz, gz8, 0);
#else
    __builtin_amdgcn_tensor_load_to_lds(g0, g1, gz, gz, 0);
#endif
}
#endif

// ---------------- WMMA fragment helpers (ISA 7.12.2 16-bit layouts) ----------

__device__ __forceinline__ v8f wmma_step(v16h a, v16h b, v8f c) {
    return __builtin_amdgcn_wmma_f32_16x16x32_f16(
        false, a, false, b, (short)0, c, false, false);
}

__device__ __forceinline__ v16h load_a_frag(const _Float16* A, int lda, int m0, int k0) {
    int lane = threadIdx.x & 31;
    int m  = m0 + (lane & 15);
    int kb = (lane < 16) ? 0 : 8;
    v16h f;
#pragma unroll
    for (int v = 0; v < 8; ++v) {
        int kbase = (v < 4) ? (kb + 2 * v) : (16 + kb + 2 * (v - 4));
        const _Float16* p = A + (size_t)m * lda + k0 + kbase;
        f[2 * v]     = p[0];
        f[2 * v + 1] = p[1];
    }
    return f;
}

__device__ __forceinline__ v16h load_b_frag(const _Float16* W, int ldb, int n0, int k0) {
    int lane = threadIdx.x & 31;
    int n  = n0 + (lane & 15);
    int kb = (lane < 16) ? 0 : 8;
    v16h f;
#pragma unroll
    for (int v = 0; v < 8; ++v) {
        int kbase = (v < 4) ? (kb + 2 * v) : (16 + kb + 2 * (v - 4));
        const _Float16* p = W + (size_t)n * ldb + k0 + kbase;
        f[2 * v]     = p[0];
        f[2 * v + 1] = p[1];
    }
    return f;
}

// ---------------- kernel 0: init / f16 weight conversion / state zero --------

__global__ void init_kernel(const float* th_w, const float* pop_w, const float* agg_w,
                            _Float16* thw_h, _Float16* popw_h, _Float16* aggw_h,
                            _Float16* sp_g, unsigned* sync) {
    const size_t NTH = (size_t)NPOP_ * NB_ * PSZ_ * C_;     // 638976
    const size_t NPW = (size_t)NCONN_ * NB_ * PSZ_ * PSZ_;  // 1477632
    const size_t NAG = (size_t)PSZ_ * CATD_;                // 8192
    const size_t NST = (size_t)NPOP_ * NB_ * B_ * PSZ_;     // 638976
    size_t stride = (size_t)gridDim.x * blockDim.x;
    for (size_t e = (size_t)blockIdx.x * blockDim.x + threadIdx.x; e < NPW; e += stride) {
        if (e < NTH) thw_h[e]  = (_Float16)th_w[e];
        popw_h[e] = (_Float16)pop_w[e];
        if (e < NAG) aggw_h[e] = (_Float16)agg_w[e];
        if (e < NST) sp_g[e] = (_Float16)0.0f;
        if (e < 8)   sync[e] = 0u;
    }
}

// ---------------- kernel 1: temporal compressor (Conv1d 2->64, k3 s2 p1) -----

__global__ void conv_kernel(const float* __restrict__ x, const float* __restrict__ cw,
                            const float* __restrict__ cb, _Float16* __restrict__ xt_h) {
    size_t gid = (size_t)blockIdx.x * blockDim.x + threadIdx.x;
    const size_t total = (size_t)B_ * NB_ * T_ * C_;
    if (gid >= total) return;
    int oc = (int)(gid & 63);
    int tt = (int)((gid >> 6) & 63);
    size_t bn = gid >> 12;
    int n = (int)(bn % NB_);
    int b = (int)(bn / NB_);
    float acc = cb[oc];
    const float* xp = x + bn * 2 * 128;
#pragma unroll
    for (int ic = 0; ic < 2; ++ic) {
#pragma unroll
        for (int kk = 0; kk < 3; ++kk) {
            int pos = 2 * tt - 1 + kk;
            if (pos >= 0 && pos < 128)
                acc += xp[ic * 128 + pos] * cw[(oc * 2 + ic) * 3 + kk];
        }
    }
    xt_h[(((size_t)tt * NB_ + n) * B_ + b) * C_ + oc] = (_Float16)acc;
}

// ---------------- kernel 2: top-3 neighbor indices per (t,b,n) ---------------

__global__ void topk_kernel(const _Float16* __restrict__ xt_h, int* __restrict__ idx_g) {
    size_t gid = (size_t)blockIdx.x * blockDim.x + threadIdx.x;
    const size_t total = (size_t)T_ * B_ * NB_;
    if (gid >= total) return;
    int n = (int)(gid % NB_);
    int b = (int)((gid / NB_) % B_);
    int t = (int)(gid / ((size_t)NB_ * B_));
    const _Float16* base = xt_h + (size_t)t * NB_ * B_ * C_;
    float xn[C_];
#pragma unroll
    for (int c = 0; c < C_; ++c)
        xn[c] = (float)base[((size_t)n * B_ + b) * C_ + c];
    float v0 = -1e30f, v1 = -1e30f, v2 = -1e30f;
    int i0 = 0, i1 = 0, i2 = 0;
    for (int m = 0; m < NB_; ++m) {
        const _Float16* xm = base + ((size_t)m * B_ + b) * C_;
        float d = 0.0f;
#pragma unroll
        for (int c = 0; c < C_; ++c) d += xn[c] * (float)xm[c];
        if (d > v0)      { v2 = v1; i2 = i1; v1 = v0; i1 = i0; v0 = d; i0 = m; }
        else if (d > v1) { v2 = v1; i2 = i1; v1 = d;  i1 = m; }
        else if (d > v2) { v2 = d;  i2 = m; }
    }
    int* op = idx_g + (((size_t)t * B_ + b) * NB_ + n) * 3;
    op[0] = i0; op[1] = i1; op[2] = i2;
}

// ---------------- global sense-reversing barrier -----------------------------

__device__ __forceinline__ void gsync(unsigned* cnt, volatile unsigned* gen, unsigned* lgen) {
    __syncthreads();
    if (threadIdx.x == 0) {
        __threadfence();
        unsigned g = *lgen;
        if (atomicAdd(cnt, 1u) == gridDim.x - 1u) {
            *cnt = 0u;
            __threadfence();
            atomicAdd((unsigned*)gen, 1u);
        } else {
            while (*gen == g) { __builtin_amdgcn_s_sleep(2); }
        }
        *lgen = g + 1u;
    }
    __syncthreads();
}

// ---------------- kernel 3: persistent recurrent scan ------------------------
// grid = 39 blocks (one barrel each), 256 threads = 8 wave32s.
// dynamic LDS: Xt (64x64 f16) | own spikes (8x64x32 f16) | nbr (64x256 f16)

__global__ void __launch_bounds__(256, 1)
recurrent_kernel(const _Float16* __restrict__ xt_h,     // (T,NB,B,64)
                 const _Float16* __restrict__ thw_h,    // (NPOP,NB,32,64)
                 const _Float16* __restrict__ popw_h,   // (NCONN,NB,32,32)
                 const _Float16* __restrict__ aggw_h,   // (32,256)
                 const float* __restrict__ th_b,        // (NPOP,NB,32)
                 const float* __restrict__ pop_b,       // (NCONN,NB,32)
                 const float* __restrict__ agg_b,       // (32)
                 const float* __restrict__ tau_adp,     // (NPOP,NB,32)
                 const float* __restrict__ tau_m,       // (NPOP,NB,32)
                 const int* __restrict__ idx_g,         // (T,B,NB,3)
                 _Float16* sp_g,                        // (NPOP,NB,B,32)
                 float* h_out, float* m_out,            // (B,T,NB,256)
                 unsigned* sync) {
    extern __shared__ char smem[];
    _Float16* sXt  = (_Float16*)smem;                        // 64*64
    _Float16* sSp  = sXt + B_ * C_;                          // 8*64*32
    _Float16* sNbr = sSp + NPOP_ * B_ * PSZ_;                // 64*256

    const int n    = blockIdx.x;
    const int tid  = threadIdx.x;
    const int lane = tid & 31;
    const int wave = tid >> 5;
    const int m0 = (wave & 3) * 16;   // batch tile
    const int n0 = (wave >> 2) * 16;  // neuron tile
    const int o  = n0 + (lane & 15);  // this lane's output neuron (fixed)
    unsigned lgen = 0u;

    // zero own-barrel spike tiles
    for (int e = tid; e < NPOP_ * B_ * PSZ_; e += 256) sSp[e] = (_Float16)0.0f;

    // per-pop invariants hoisted out of the time loop (registers)
    float biasv[NPOP_], alphav[NPOP_], rov[NPOP_];
#pragma unroll
    for (int i = 0; i < NPOP_; ++i) {
        size_t po = ((size_t)i * NB_ + n) * PSZ_ + o;
        float bs = th_b[po];
#pragma unroll
        for (int jj = 0; jj < kNSRC[i]; ++jj)
            bs += pop_b[((size_t)(kOFF[i] + jj) * NB_ + n) * PSZ_ + o];
        biasv[i]  = bs;
        alphav[i] = __expf(-1.0f / tau_m[po]);
        rov[i]    = __expf(-1.0f / tau_adp[po]);
    }
    const float aggb = agg_b[o] * 0.1f;

    // agg_w B-fragments: shared by every pop-step -> load once
    v16h aggB[8];
#pragma unroll
    for (int ks = 0; ks < 8; ++ks) aggB[ks] = load_b_frag(aggw_h, CATD_, n0, ks * 32);

    // LIF state in registers (static indexing after full unroll)
    float memv[NPOP_][8] = {};
    float bbv[NPOP_][8]  = {};

    __syncthreads();

    for (int t = 0; t < T_; ++t) {
        // stage Xt(64x64 f16) for this (t, barrel) into LDS
        const _Float16* xsrc = xt_h + ((size_t)t * NB_ + n) * B_ * C_;
#if HAVE_TDM
        if (wave == 0) {                      // one TDM descriptor per block
            tdm_load_1d(0u, xsrc, B_ * C_);
            __builtin_amdgcn_s_wait_tensorcnt(0);
        }
#else
        {
            const unsigned* src = (const unsigned*)xsrc;
            unsigned* dst = (unsigned*)sXt;
            for (int e = tid; e < B_ * C_ / 2; e += 256) dst[e] = src[e];
        }
#endif
        __syncthreads();

#pragma unroll
        for (int i = 0; i < NPOP_; ++i) {
            // gather mean of top-3 neighbor concat-spikes (packed 2xf16)
            if (t > 0) {
                int gb = tid >> 2;            // batch row, 4 threads per row
                int gq = tid & 3;
                const int* ip = idx_g + (((size_t)t * B_ + gb) * NB_ + n) * 3;
                int mi0 = ip[0], mi1 = ip[1], mi2 = ip[2];
                const unsigned* spu = (const unsigned*)sp_g;
                h2* nrow = (h2*)sNbr + (size_t)gb * 128;
#pragma unroll 8
                for (int u = 0; u < 32; ++u) {
                    int c2 = gq * 32 + u;     // u32 channel index [0,128)
                    int p  = c2 >> 4;
                    int o2 = c2 & 15;
                    unsigned w0 = spu[(((size_t)p * NB_ + mi0) * B_ + gb) * 16 + o2];
                    unsigned w1 = spu[(((size_t)p * NB_ + mi1) * B_ + gb) * 16 + o2];
                    unsigned w2 = spu[(((size_t)p * NB_ + mi2) * B_ + gb) * 16 + o2];
                    h2 p0 = __builtin_bit_cast(h2, w0);
                    h2 p1 = __builtin_bit_cast(h2, w1);
                    h2 p2 = __builtin_bit_cast(h2, w2);
                    h2 rv;
                    rv[0] = (_Float16)(((float)p0[0] + (float)p1[0] + (float)p2[0]) * (1.0f / 3.0f));
                    rv[1] = (_Float16)(((float)p0[1] + (float)p1[1] + (float)p2[1]) * (1.0f / 3.0f));
                    nrow[c2] = rv;
                }
            }
            __syncthreads();

            if (i + 1 < NPOP_)
                __builtin_prefetch((const void*)(thw_h + ((size_t)(i + 1) * NB_ + n) * PSZ_ * C_), 0, 0);

            // ---- WMMA GEMMs: cur(64x32), tile (m0,n0) per wave ----
            v8f acc = {};
            const _Float16* thw = thw_h + ((size_t)i * NB_ + n) * PSZ_ * C_;
#pragma unroll
            for (int ks = 0; ks < 2; ++ks) {                 // thalamic drive, K=64
                v16h a = load_a_frag(sXt, C_, m0, ks * 32);
                v16h b = load_b_frag(thw, C_, n0, ks * 32);
                acc = wmma_step(a, b, acc);
            }
#pragma unroll
            for (int jj = 0; jj < kNSRC[i]; ++jj) {          // recurrent pops, K=32
                constexpr const int* srcp = kSRC;            // folded
                int k = kOFF[i] + jj;
                int j = srcp[kOFF[i] + jj];
                v16h a = load_a_frag(sSp + (size_t)j * B_ * PSZ_, PSZ_, m0, 0);
                v16h b = load_b_frag(popw_h + ((size_t)k * NB_ + n) * PSZ_ * PSZ_, PSZ_, n0, 0);
                acc = wmma_step(a, b, acc);
            }
            v8f accAgg = {};
            if (t > 0) {
#pragma unroll
                for (int ks = 0; ks < 8; ++ks) {             // neighbor agg, K=256
                    v16h a = load_a_frag(sNbr, CATD_, m0, ks * 32);
                    accAgg = wmma_step(a, aggB[ks], accAgg);
                }
            }
            __syncthreads();   // reads of sSp[i] complete before update writes

            // ---- adaptive LIF update (state in registers) ----
#pragma unroll
            for (int r = 0; r < 8; ++r) {
                int b = m0 + ((lane >> 4) << 3) + r;
                float cur = acc[r] + biasv[i];
                if (t > 0) cur += 0.1f * accAgg[r] + aggb;
                float mem = memv[i][r];
                float bb  = bbv[i][r];
                float spko = (float)sSp[((size_t)i * B_ + b) * PSZ_ + o];
                bb = rov[i] * bb + (1.0f - rov[i]) * spko;
                float Bth = 0.02f + 1.8f * bb;
                mem = mem * alphav[i] + (1.0f - alphav[i]) * cur;
                float s = (mem - Bth) > 0.0f ? 1.0f : 0.0f;
                mem = mem - Bth * s;
                memv[i][r] = mem;
                bbv[i][r]  = bb;
                _Float16 sh = (_Float16)s;
                sSp[((size_t)i * B_ + b) * PSZ_ + o] = sh;
                sp_g[(((size_t)i * NB_ + n) * B_ + b) * PSZ_ + o] = sh;
                size_t oidx = (((size_t)b * T_ + t) * NB_ + n) * CATD_ + i * PSZ_ + o;
                h_out[oidx] = s;
                m_out[oidx] = mem + s * Bth;
            }
            __threadfence();
            gsync(&sync[0], &sync[1], &lgen);   // publish spikes to all barrels
        }
    }
}

// ---------------- kernel 4: readout (mean over t,n of L5 -> linear -> softmax)

__global__ void readout_kernel(const float* __restrict__ h_state,
                               const float* __restrict__ ro_w,
                               const float* __restrict__ ro_b,
                               float* __restrict__ out) {
    int b = blockIdx.x;
    int tid = threadIdx.x;
    __shared__ float ssum[256];
    __shared__ float cmean[128];
    __shared__ float logits[36];
    int c = tid & 127;
    int half = tid >> 7;
    float s = 0.0f;
    for (int e = half; e < T_ * NB_; e += 2) {
        int t = e / NB_;
        int n = e % NB_;
        s += h_state[(((size_t)b * T_ + t) * NB_ + n) * CATD_ + 128 + c];
    }
    ssum[tid] = s;
    __syncthreads();
    if (tid < 128) cmean[tid] = (ssum[tid] + ssum[tid + 128]) * (1.0f / (float)(T_ * NB_));
    __syncthreads();
    if (tid < 36) {
        float acc = ro_b[tid];
        for (int c2 = 0; c2 < 128; ++c2) acc += cmean[c2] * ro_w[tid * 128 + c2];
        logits[tid] = acc;
    }
    __syncthreads();
    if (tid == 0) {
        float mx = logits[0];
        for (int j = 1; j < 36; ++j) mx = fmaxf(mx, logits[j]);
        float sum = 0.0f;
        for (int j = 0; j < 36; ++j) sum += __expf(logits[j] - mx);
        float inv = 1.0f / sum;
        for (int j = 0; j < 36; ++j) out[(size_t)b * 36 + j] = __expf(logits[j] - mx) * inv;
    }
}

// ---------------- host-side launch -------------------------------------------

extern "C" void kernel_launch(void* const* d_in, const int* in_sizes, int n_in,
                              void* d_out, int out_size, void* d_ws, size_t ws_size,
                              hipStream_t stream) {
    const float* inputs  = (const float*)d_in[0];
    const float* comp_w  = (const float*)d_in[1];
    const float* comp_b  = (const float*)d_in[2];
    const float* th_w    = (const float*)d_in[3];
    const float* th_b    = (const float*)d_in[4];
    const float* pop_w   = (const float*)d_in[5];
    const float* pop_b   = (const float*)d_in[6];
    const float* tau_adp = (const float*)d_in[7];
    const float* tau_m   = (const float*)d_in[8];
    const float* agg_w   = (const float*)d_in[9];
    const float* agg_b   = (const float*)d_in[10];
    const float* ro_w    = (const float*)d_in[11];
    const float* ro_b    = (const float*)d_in[12];

    (void)in_sizes; (void)n_in; (void)out_size; (void)ws_size;

    char* ws = (char*)d_ws;
    const size_t SZ_XT   = (size_t)T_ * NB_ * B_ * C_ * 2;         // 20,447,232
    const size_t SZ_THW  = (size_t)NPOP_ * NB_ * PSZ_ * C_ * 2;    //  1,277,952
    const size_t SZ_POPW = (size_t)NCONN_ * NB_ * PSZ_ * PSZ_ * 2; //  2,955,264
    const size_t SZ_AGGW = (size_t)PSZ_ * CATD_ * 2;               //     16,384
    const size_t SZ_SPG  = (size_t)NPOP_ * NB_ * B_ * PSZ_ * 2;    //  1,277,952
    const size_t SZ_IDX  = (size_t)T_ * B_ * NB_ * 3 * 4;          //  1,916,928

    _Float16* xt_h   = (_Float16*)(ws);
    _Float16* thw_h  = (_Float16*)(ws + SZ_XT);
    _Float16* popw_h = (_Float16*)(ws + SZ_XT + SZ_THW);
    _Float16* aggw_h = (_Float16*)(ws + SZ_XT + SZ_THW + SZ_POPW);
    _Float16* sp_g   = (_Float16*)(ws + SZ_XT + SZ_THW + SZ_POPW + SZ_AGGW);
    int*      idx_g  = (int*)     (ws + SZ_XT + SZ_THW + SZ_POPW + SZ_AGGW + SZ_SPG);
    unsigned* sync   = (unsigned*)(ws + SZ_XT + SZ_THW + SZ_POPW + SZ_AGGW + SZ_SPG + SZ_IDX);

    float* out_p = (float*)d_out;                         // (64,36)
    float* h_out = out_p + (size_t)B_ * 36;               // (64,64,39,256)
    float* m_out = h_out + (size_t)B_ * T_ * NB_ * CATD_; // (64,64,39,256)

    init_kernel<<<1024, 256, 0, stream>>>(th_w, pop_w, agg_w, thw_h, popw_h, aggw_h,
                                          sp_g, sync);
    {
        size_t total = (size_t)B_ * NB_ * T_ * C_;
        conv_kernel<<<(unsigned)((total + 255) / 256), 256, 0, stream>>>(inputs, comp_w, comp_b, xt_h);
    }
    {
        size_t total = (size_t)T_ * B_ * NB_;
        topk_kernel<<<(unsigned)((total + 255) / 256), 256, 0, stream>>>(xt_h, idx_g);
    }

    const size_t lds_bytes = (size_t)(B_ * C_ + NPOP_ * B_ * PSZ_ + B_ * CATD_) * sizeof(_Float16); // 73728
    recurrent_kernel<<<NB_, 256, lds_bytes, stream>>>(xt_h, thw_h, popw_h, aggw_h,
                                                      th_b, pop_b, agg_b, tau_adp, tau_m,
                                                      idx_g, sp_g, h_out, m_out, sync);

    readout_kernel<<<B_, 256, 0, stream>>>(h_out, ro_w, ro_b, out_p);
}
